// NSAAttention_87625922773704
// MI455X (gfx1250) — matrix operand order repaired and checked by reading, compiled, not run on hardware
//
#include <hip/hip_runtime.h>
#include <hip/hip_bf16.h>
#include <math.h>

// ---------------- problem constants (from reference) ----------------
#define B_      2
#define S_      2048
#define D_      1024
#define H_      16
#define DH_     64
#define CBS_    32
#define STRIDE_ 16
#define NSEL_   16
#define WIN_    16
#define NBLK_   127            // (S - CBS)/STRIDE + 1
#define NW_     33             // window positions (2*WIN+1)
#define NSC_    49             // 33 local + 16 global scores
#define SCALE_  0.125f         // DH^-0.5

// ---------------- WMMA vector types (CDNA5, wave32) ----------------
typedef __attribute__((ext_vector_type(16))) __bf16 v16bf;
typedef __attribute__((ext_vector_type(8)))  __bf16 v8bf;
typedef __attribute__((ext_vector_type(8)))  float  v8f;

union BF16Frag { v16bf v; v8bf h[2]; };

#define WMMA_BF16(a, b, c) \
  __builtin_amdgcn_wmma_f32_16x16x32_bf16(false, (a), false, (b), (short)0, (c), false, false)

// ---------------- generic bf16 WMMA GEMM ----------------
// C[M,N](f32) = act( A[M,K](bf16, optional per-row element offsets) @ Bt^T + bias )
// Workgroup: 256 threads = 8 waves, tile 128(M) x 128(N); wave tile 32x64.
// Requires: N % 128 == 0, K % 32 == 0 (true for all uses here).
//
// All per-lane base pointers are hoisted out of the K-loop so the steady state
// is 12x global_load_b128 + 8x v_wmma per K=32 step (no branches, no scalar
// reloads inside the loop).
__global__ __launch_bounds__(256) void gemm_bf16_wmma(
    const __bf16* __restrict__ A, const int* __restrict__ arow_off, int lda,
    const __bf16* __restrict__ Bt, const float* __restrict__ bias,
    float* __restrict__ C, int ldc, int M, int N, int K, int relu)
{
  const int lane = threadIdx.x & 31;
  const int wave = threadIdx.x >> 5;
  const int m0 = blockIdx.x * 128 + (wave & 3) * 32;
  const int n0 = blockIdx.y * 128 + (wave >> 2) * 64;

  // ---- hoisted per-lane A row pointers (A-frag, contiguous k order k = 16h + s) ----
  // Lane L: row M = m0 + (L&15); per-lane K phase kb = (L>=16)*16; the second
  // 8 elements come from +8.
  int ma = m0 + (lane & 15);       if (ma >= M) ma = M - 1;   // clamp; store guarded
  int mb = ma + 16;                 if (mb >= M) mb = M - 1;
  const long ba = arow_off ? (long)arow_off[ma] : (long)ma * lda;
  const long bb = arow_off ? (long)arow_off[mb] : (long)mb * lda;
  const int  kphA = (lane >> 4) << 4;     // 0 or 16
  const __bf16* __restrict__ ap0 = A + ba + kphA;
  const __bf16* __restrict__ ap1 = A + bb + kphA;

  // ---- hoisted per-lane B row pointers (B-frag, contiguous k order k = 16h + s) ----
  // Lane L: column N = n0 + (L&15); K phase (L>=16)*16; 16 contiguous bf16.
  const long  bn  = (long)(n0 + (lane & 15)) * K + ((lane >> 4) << 4);
  const __bf16* __restrict__ bp0 = Bt + bn;
  const __bf16* __restrict__ bp1 = bp0 + 16L * K;
  const __bf16* __restrict__ bp2 = bp0 + 32L * K;
  const __bf16* __restrict__ bp3 = bp0 + 48L * K;

  v8f zero = {};
  v8f acc[2][4];
#pragma unroll
  for (int i = 0; i < 2; ++i)
#pragma unroll
    for (int j = 0; j < 4; ++j) acc[i][j] = zero;

  for (int k0 = 0; k0 < K; k0 += 32) {
    BF16Frag a0, a1, b0, b1, b2, b3;
    a0.h[0] = *(const v8bf*)(ap0 + k0);
    a0.h[1] = *(const v8bf*)(ap0 + k0 + 8);
    a1.h[0] = *(const v8bf*)(ap1 + k0);
    a1.h[1] = *(const v8bf*)(ap1 + k0 + 8);
    b0.h[0] = *(const v8bf*)(bp0 + k0);
    b0.h[1] = *(const v8bf*)(bp0 + k0 + 8);
    b1.h[0] = *(const v8bf*)(bp1 + k0);
    b1.h[1] = *(const v8bf*)(bp1 + k0 + 8);
    b2.h[0] = *(const v8bf*)(bp2 + k0);
    b2.h[1] = *(const v8bf*)(bp2 + k0 + 8);
    b3.h[0] = *(const v8bf*)(bp3 + k0);
    b3.h[1] = *(const v8bf*)(bp3 + k0 + 8);

    if (k0 + 32 < K) {  // hint next K chunk into cache (global_prefetch_b8)
      __builtin_prefetch((const void*)(ap0 + k0 + 32), 0, 0);
      __builtin_prefetch((const void*)(bp0 + k0 + 32), 0, 0);
    }

    acc[0][0] = WMMA_BF16(a0.v, b0.v, acc[0][0]);
    acc[0][1] = WMMA_BF16(a0.v, b1.v, acc[0][1]);
    acc[0][2] = WMMA_BF16(a0.v, b2.v, acc[0][2]);
    acc[0][3] = WMMA_BF16(a0.v, b3.v, acc[0][3]);
    acc[1][0] = WMMA_BF16(a1.v, b0.v, acc[1][0]);
    acc[1][1] = WMMA_BF16(a1.v, b1.v, acc[1][1]);
    acc[1][2] = WMMA_BF16(a1.v, b2.v, acc[1][2]);
    acc[1][3] = WMMA_BF16(a1.v, b3.v, acc[1][3]);
  }

  // C/D layout: lane L holds column n0+(L&15); VGPR j holds row (L>>4)*8 + j.
#pragma unroll
  for (int mi = 0; mi < 2; ++mi) {
#pragma unroll
    for (int ni = 0; ni < 4; ++ni) {
      int n = n0 + ni * 16 + (lane & 15);
      float bv = bias ? bias[n] : 0.f;
#pragma unroll
      for (int j = 0; j < 8; ++j) {
        int m = m0 + mi * 16 + ((lane >> 4) << 3) + j;
        if (m < M) {
          float v = acc[mi][ni][j] + bv;
          if (relu) v = fmaxf(v, 0.f);
          C[(long)m * ldc + n] = v;
        }
      }
    }
  }
}

// ---------------- f32 -> bf16 convert ----------------
__global__ void cvt_f32_to_bf16(const float* __restrict__ in, __bf16* __restrict__ out, long n) {
  long i = (long)blockIdx.x * blockDim.x + threadIdx.x;
  long stride = (long)gridDim.x * blockDim.x;
  for (; i < n; i += stride) out[i] = (__bf16)in[i];
}

// ---------------- f32 [K,N] -> bf16 [N,K] transpose-convert ----------------
__global__ __launch_bounds__(256) void transpose_cvt(const float* __restrict__ W,
                                                     __bf16* __restrict__ Wt, int K, int N) {
  __shared__ float t[32][33];
  int n0 = blockIdx.x * 32, k0 = blockIdx.y * 32;
  for (int i = threadIdx.y; i < 32; i += 8) {
    int k = k0 + i, n = n0 + threadIdx.x;
    t[i][threadIdx.x] = (k < K && n < N) ? W[(long)k * N + n] : 0.f;
  }
  __syncthreads();
  for (int i = threadIdx.y; i < 32; i += 8) {
    int n = n0 + i, k = k0 + threadIdx.x;
    if (n < N && k < K) Wt[(long)n * K + k] = (__bf16)t[threadIdx.x][i];
  }
}

// ---------------- row-offset table for the overlapping compression blocks ----
__global__ void make_crow_off(int* __restrict__ off) {
  int r = blockIdx.x * blockDim.x + threadIdx.x;
  if (r < B_ * NBLK_) {
    int b = r / NBLK_, i = r % NBLK_;
    off[r] = (b * S_ + i * STRIDE_) * D_;   // block row = 32 contiguous x rows
  }
}

// ---------------- per-row score: compressed @ score_w + score_b -------------
__global__ __launch_bounds__(256) void score_rows(const float* __restrict__ comp,
                                                  const float* __restrict__ sw,
                                                  const float* __restrict__ sb,
                                                  float* __restrict__ scores, int M) {
  int wave = threadIdx.x >> 5, lane = threadIdx.x & 31;
  int r = blockIdx.x * 8 + wave;
  if (r >= M) return;
  const float* row = comp + (long)r * D_;
  float acc = 0.f;
  for (int d = lane; d < D_; d += 32) acc += row[d] * sw[d];
#pragma unroll
  for (int m = 16; m; m >>= 1) acc += __shfl_xor(acc, m, 32);
  if (lane == 0) scores[r] = acc + sb[0];
}

// ---------------- top-16 per batch (one wave per batch) ---------------------
__global__ __launch_bounds__(32) void topk_sel(const float* __restrict__ scores,
                                               int* __restrict__ sel_off) {
  int b = blockIdx.x;
  int lane = threadIdx.x;
  __shared__ float s[128];
  for (int i = lane; i < NBLK_; i += 32) s[i] = scores[b * NBLK_ + i];
  if (lane == 0) s[NBLK_] = -INFINITY;
  __syncthreads();
  for (int sel = 0; sel < NSEL_; ++sel) {
    float best = -INFINITY; int bi = NBLK_;
    for (int i = lane; i < NBLK_; i += 32) {
      float v = s[i];
      if (v > best || (v == best && i < bi)) { best = v; bi = i; }
    }
#pragma unroll
    for (int m = 16; m; m >>= 1) {
      float ob = __shfl_xor(best, m, 32);
      int   oi = __shfl_xor(bi,   m, 32);
      if (ob > best || (ob == best && oi < bi)) { best = ob; bi = oi; }
    }
    if (lane == 0) {
      sel_off[b * NSEL_ + sel] = (b * NBLK_ + bi) * D_;  // element offset into compressed
      s[bi] = -INFINITY;
    }
    __syncthreads();
  }
}

// ---------------- fused local(window) + global(selected) attention ----------
// One wave per (b, h, s). qkvf rows: [q(0..1023) | k(1024..2047) | v(2048..3071)].
__global__ __launch_bounds__(256) void nsa_attention(const float* __restrict__ qkvf,
                                                     const float* __restrict__ selqkv,
                                                     float* __restrict__ attn) {
  const int wave = threadIdx.x >> 5, lane = threadIdx.x & 31;
  const int item = blockIdx.x * 8 + wave;       // exactly B*H*S items
  const int b = item / (H_ * S_);
  const int rem = item % (H_ * S_);
  const int h = rem / S_;
  const int s = rem % S_;

  __shared__ float sc[8][NSC_];

  const float* qp = qkvf + (long)(b * S_ + s) * (3 * D_) + h * DH_;

  for (int idx = lane; idx < NSC_; idx += 32) {
    float dot = 0.f;
    if (idx < NW_) {
      int pos = s - WIN_ + idx;
      int posc = pos < 0 ? 0 : (pos >= S_ ? S_ - 1 : pos);
      const float* kp = qkvf + (long)(b * S_ + posc) * (3 * D_) + D_ + h * DH_;
      for (int d = 0; d < DH_; ++d) dot += qp[d] * kp[d];
      dot *= SCALE_;
      if (pos < 0 || pos >= S_) dot = -INFINITY;
    } else {
      int j = idx - NW_;
      const float* kp = selqkv + (long)(b * NSEL_ + j) * (3 * D_) + D_ + h * DH_;
      for (int d = 0; d < DH_; ++d) dot += qp[d] * kp[d];
      dot *= SCALE_;
    }
    sc[wave][idx] = dot;
  }
  __syncthreads();

  float m1 = -INFINITY, m2 = -INFINITY;
  for (int i = 0; i < NW_; ++i)        m1 = fmaxf(m1, sc[wave][i]);
  for (int i = NW_; i < NSC_; ++i)     m2 = fmaxf(m2, sc[wave][i]);
  float s1 = 0.f, s2 = 0.f;
  for (int i = 0; i < NW_; ++i)        s1 += expf(sc[wave][i] - m1);
  for (int i = NW_; i < NSC_; ++i)     s2 += expf(sc[wave][i] - m2);
  const float r1 = 1.f / s1, r2 = 1.f / s2;

  float accLo = 0.f, accHi = 0.f;
  for (int i = 0; i < NW_; ++i) {
    float p = expf(sc[wave][i] - m1) * r1;      // masked -> exp(-inf)=0
    int pos = s - WIN_ + i;
    int posc = pos < 0 ? 0 : (pos >= S_ ? S_ - 1 : pos);
    const float* vp = qkvf + (long)(b * S_ + posc) * (3 * D_) + 2 * D_ + h * DH_;
    accLo += p * vp[lane];
    accHi += p * vp[lane + 32];
  }
  for (int j = 0; j < NSEL_; ++j) {
    float g = expf(sc[wave][NW_ + j] - m2) * r2;
    const float* vp = selqkv + (long)(b * NSEL_ + j) * (3 * D_) + 2 * D_ + h * DH_;
    accLo += g * vp[lane];
    accHi += g * vp[lane + 32];
  }

  float* op = attn + (long)(b * S_ + s) * D_ + h * DH_;
  op[lane]      = accLo;
  op[lane + 32] = accHi;
}

// ---------------- host orchestration ----------------
extern "C" void kernel_launch(void* const* d_in, const int* in_sizes, int n_in,
                              void* d_out, int out_size, void* d_ws, size_t ws_size,
                              hipStream_t stream) {
  const float* x       = (const float*)d_in[0];   // (B,S,D)
  const float* qkv_w   = (const float*)d_in[1];   // (D, 3D)
  const float* qkv_b   = (const float*)d_in[2];   // (3D,)
  const float* c1_w    = (const float*)d_in[3];   // (CBS*D, D)
  const float* c1_b    = (const float*)d_in[4];   // (D,)
  const float* c2_w    = (const float*)d_in[5];   // (D, D)
  const float* c2_b    = (const float*)d_in[6];   // (D,)
  const float* score_w = (const float*)d_in[7];   // (D, 1)
  const float* score_b = (const float*)d_in[8];   // (1,)
  const float* out_w   = (const float*)d_in[9];   // (D, D)
  const float* out_b   = (const float*)d_in[10];  // (D,)
  float* outp = (float*)d_out;                    // (B,S,D) f32

  const int BS  = B_ * S_;        // 4096
  const int M2  = B_ * NBLK_;     // 254
  const int K1  = CBS_ * D_;      // 32768

  // workspace carve-up (256B aligned)
  char* base = (char*)d_ws;
  size_t off = 0;
  auto alloc = [&](size_t bytes) -> void* {
    void* p = base + off;
    off = (off + bytes + 255) & ~(size_t)255;
    return p;
  };
  __bf16* xb      = (__bf16*)alloc((size_t)BS * D_ * 2);
  __bf16* qkv_wt  = (__bf16*)alloc((size_t)3 * D_ * D_ * 2);       // (3D, D)
  __bf16* c1_wt   = (__bf16*)alloc((size_t)D_ * K1 * 2);           // (D, 32768)
  __bf16* c2_wt   = (__bf16*)alloc((size_t)D_ * D_ * 2);
  __bf16* out_wt  = (__bf16*)alloc((size_t)D_ * D_ * 2);
  float*  qkvf    = (float*)alloc((size_t)BS * 3 * D_ * 4);
  float*  hidden  = (float*)alloc((size_t)M2 * D_ * 4);
  __bf16* hiddenb = (__bf16*)alloc((size_t)M2 * D_ * 2);
  float*  comp    = (float*)alloc((size_t)M2 * D_ * 4);
  __bf16* compb   = (__bf16*)alloc((size_t)M2 * D_ * 2);
  float*  scores  = (float*)alloc((size_t)M2 * 4);
  int*    crow    = (int*)alloc((size_t)M2 * 4);
  int*    seloff  = (int*)alloc((size_t)B_ * NSEL_ * 4);
  float*  selqkv  = (float*)alloc((size_t)B_ * NSEL_ * 3 * D_ * 4);
  float*  attn    = (float*)alloc((size_t)BS * D_ * 4);
  __bf16* attnb   = (__bf16*)alloc((size_t)BS * D_ * 2);

  auto cvt = [&](const float* in, __bf16* out2, long n) {
    long blocks = (n + 255) / 256; if (blocks > 8192) blocks = 8192;
    hipLaunchKernelGGL(cvt_f32_to_bf16, dim3((unsigned)blocks), dim3(256), 0, stream, in, out2, n);
  };
  auto tcvt = [&](const float* W, __bf16* Wt, int K, int N) {
    hipLaunchKernelGGL(transpose_cvt, dim3(N / 32, K / 32), dim3(32, 8), 0, stream, W, Wt, K, N);
  };
  auto gemm = [&](const __bf16* A, const int* aoff, int lda, const __bf16* Bt,
                  const float* bias, float* C, int M, int N, int K, int relu) {
    dim3 g((M + 127) / 128, N / 128);
    hipLaunchKernelGGL(gemm_bf16_wmma, g, dim3(256), 0, stream,
                       A, aoff, lda, Bt, bias, C, N, M, N, K, relu);
  };

  // 1) precision conversion + weight transposes (bf16 c1_w = 64MB -> L2-resident)
  cvt(x, xb, (long)BS * D_);
  tcvt(qkv_w, qkv_wt, D_, 3 * D_);
  tcvt(c1_w,  c1_wt,  K1, D_);
  tcvt(c2_w,  c2_wt,  D_, D_);
  tcvt(out_w, out_wt, D_, D_);

  // 2) QKV projection: (4096 x 1024) @ (1024 x 3072)
  gemm(xb, nullptr, D_, qkv_wt, qkv_b, qkvf, BS, 3 * D_, D_, 0);

  // 3) compression: overlapping blocks are contiguous x slices -> row-offset GEMM
  hipLaunchKernelGGL(make_crow_off, dim3(1), dim3(256), 0, stream, crow);
  gemm(xb, crow, 0, c1_wt, c1_b, hidden, M2, D_, K1, 1);          // relu
  cvt(hidden, hiddenb, (long)M2 * D_);
  gemm(hiddenb, nullptr, D_, c2_wt, c2_b, comp, M2, D_, D_, 0);
  cvt(comp, compb, (long)M2 * D_);

  // 4) scores + top-16 per batch
  hipLaunchKernelGGL(score_rows, dim3((M2 + 7) / 8), dim3(256), 0, stream,
                     comp, score_w, score_b, scores, M2);
  hipLaunchKernelGGL(topk_sel, dim3(B_), dim3(32), 0, stream, scores, seloff);

  // 5) selected rows -> QKV (gathered-row GEMM, M=32)
  gemm(compb, seloff, 0, qkv_wt, qkv_b, selqkv, B_ * NSEL_, 3 * D_, D_, 0);

  // 6) fused local-window + global-selected attention
  hipLaunchKernelGGL(nsa_attention, dim3(B_ * H_ * S_ / 8), dim3(256), 0, stream,
                     qkvf, selqkv, attn);

  // 7) output projection -> d_out
  cvt(attn, attnb, (long)BS * D_);
  gemm(attnb, nullptr, D_, out_wt, out_b, outp, BS, D_, D_, 0);
}
